// DebiasedAttention_66872640799274
// MI455X (gfx1250) — matrix operand order repaired and checked by reading, not compiled, not run on hardware
//
#include <hip/hip_runtime.h>
#include <hip/hip_bf16.h>
#include <math.h>

// CDNA5 / gfx1250 wave32 WMMA types
typedef __attribute__((ext_vector_type(16))) __bf16 v16bf;
typedef __attribute__((ext_vector_type(8)))  float  v8f;

#define B_ 4
#define H_ 16
#define S_ 1024
#define D_ 64

// LDS layout (bytes)
#define SC_OFF   0                        // 16 x 1024 f32: raw scores, then probs
#define SC_SZ    (16 * S_ * 4)
#define PART_OFF (SC_OFF + SC_SZ)         // 8 waves x 16 x 64 f32 PV partials
#define PART_SZ  (8 * 16 * 64 * 4)
#define RED_OFF  (PART_OFF + PART_SZ)     // 16 x 16 f32 reduction scratch
#define RED_SZ   (16 * 16 * 4)
#define RMAX_OFF (RED_OFF + RED_SZ)       // 16 f32 row max
#define RSUM_OFF (RMAX_OFF + 16 * 4)      // 16 f32 row sum
#define LDS_TOTAL (RSUM_OFF + 16 * 4)

// prop^(-0.1) for prop > 0: 3 VALU ops (v_log_f32, v_mul, v_exp_f32)
// instead of the ~60-op precise powf expansion.
__device__ __forceinline__ float prop_factor(float p) {
    return __builtin_amdgcn_exp2f(-0.1f * __builtin_amdgcn_logf(p));
}

__global__ __launch_bounds__(256)
void debiased_attn_kernel(const float* __restrict__ q,
                          const float* __restrict__ k,
                          const float* __restrict__ v,
                          const float* __restrict__ prop,
                          const int*   __restrict__ mask,
                          float* __restrict__ out,
                          float* __restrict__ pattn)
{
    extern __shared__ char smem[];
    float* sc   = (float*)(smem + SC_OFF);
    float* part = (float*)(smem + PART_OFF);
    float* red  = (float*)(smem + RED_OFF);
    float* rmax = (float*)(smem + RMAX_OFF);
    float* rsum = (float*)(smem + RSUM_OFF);

    const int mt   = blockIdx.x;           // 16-row tile index within S
    const int h    = blockIdx.y;
    const int b    = blockIdx.z;
    const int tid  = threadIdx.x;
    const int lane = tid & 31;
    const int wv   = tid >> 5;             // wave id 0..7
    const int half = lane >> 4;            // 0: lanes 0-15, 1: lanes 16-31
    const int l16  = lane & 15;

    const size_t head = (size_t)(b * H_ + h);
    const float* qh = q + head * S_ * D_;
    const float* kh = k + head * S_ * D_;
    const float* vh = v + head * S_ * D_;

    // ---- Build Q A-fragments once (rows mt*16..+15, two K=32 chunks of D) ----
    // A layout (16-bit, 16x32): K = kb..kb+15 with kb=half*16
    const int qrow = mt * 16 + l16;
    const float* qr = qh + (size_t)qrow * D_;
    v16bf a0{}, a1{};
    {
        const int kb = half * 16;
#pragma unroll
        for (int t = 0; t < 8; ++t) {
            a0[t]     = (__bf16)qr[kb + t];
            a0[t + 8] = (__bf16)qr[kb + 8 + t];
            a1[t]     = (__bf16)qr[32 + kb + t];
            a1[t + 8] = (__bf16)qr[32 + kb + 8 + t];
        }
    }

    // ---- Phase 1: scores = (Q K^T) * 0.125 / prop^0.1, masked -> LDS ----
    // Each wave handles column tiles jt = wv + 8*tt
    for (int tt = 0; tt < 8; ++tt) {
        const int jt = wv + tt * 8;
        const int j  = jt * 16 + l16;                 // this lane's column (B N-index)
        const float f = 0.125f * prop_factor(prop[b * S_ + j]);
        const float* kr = kh + (size_t)j * D_;
        // Prefetch next tile's K row into cache (global_prefetch_b8)
        {
            const int jn = ((wv + ((tt + 1) & 7) * 8) * 16) + l16;
            __builtin_prefetch(kh + (size_t)jn * D_, 0, 3);
        }
        // B layout (16-bit, 32x16): lane = N = column j; K rows = half*16 + 0..15
        v16bf b0{}, b1{};
#pragma unroll
        for (int t = 0; t < 16; ++t) {
            b0[t] = (__bf16)kr[half * 16 + t];
            b1[t] = (__bf16)kr[32 + half * 16 + t];
        }
        v8f acc = {};
        acc = __builtin_amdgcn_wmma_f32_16x16x32_bf16(false, a0, false, b0,
                                                      (short)0, acc, false, false);
        acc = __builtin_amdgcn_wmma_f32_16x16x32_bf16(false, a1, false, b1,
                                                      (short)0, acc, false, false);
        // C/D layout: lane holds column j, rows i = half*8 + e (e = VGPR index)
#pragma unroll
        for (int e = 0; e < 8; ++e) {
            const int i  = mt * 16 + half * 8 + e;     // global row
            const int mv = mask[((size_t)b * S_ + i) * S_ + j];
            const float s = mv ? acc[e] * f : -1e9f;
            sc[(half * 8 + e) * S_ + j] = s;
        }
    }
    __syncthreads();

    // ---- Phase 2: exact row max and sum(exp) from LDS ----
    {
        const int r = tid >> 4;            // row 0..15
        const int c = tid & 15;            // 16 threads per row
        const float* row = sc + r * S_ + c * 64;
        float mx = -INFINITY;
        for (int t = 0; t < 64; ++t) mx = fmaxf(mx, row[t]);
        red[r * 16 + c] = mx;
        __syncthreads();
        if (tid < 16) {
            float m = -INFINITY;
            for (int c2 = 0; c2 < 16; ++c2) m = fmaxf(m, red[tid * 16 + c2]);
            rmax[tid] = m;
        }
        __syncthreads();
        const float rm = rmax[r];
        float s = 0.f;
        for (int t = 0; t < 64; ++t) s += __expf(row[t] - rm);
        red[r * 16 + c] = s;
        __syncthreads();
        if (tid < 16) {
            float s2 = 0.f;
            for (int c2 = 0; c2 < 16; ++c2) s2 += red[tid * 16 + c2];
            rsum[tid] = s2;
        }
        __syncthreads();
    }

    // ---- Phase 3: normalized probs -> p_attn (single coalesced write) + LDS ----
    {
        float* pbase = pattn + (head * S_ + (size_t)mt * 16) * S_;
        for (int rr = 0; rr < 16; ++rr) {
            const float rm  = rmax[rr];
            const float inv = 1.0f / rsum[rr];
            float* srow = sc + rr * S_;
            float* grow = pbase + (size_t)rr * S_;
            for (int j = tid; j < S_; j += 256) {
                const float p = __expf(srow[j] - rm) * inv;
                grow[j] = p;
                srow[j] = p;                 // overwrite score with prob for PV
            }
        }
    }
    __syncthreads();

    // ---- Phase 4: out_partial = P_chunk . V_chunk, chunk = 128 cols per wave ----
    v8f oacc[4] = {v8f{}, v8f{}, v8f{}, v8f{}};
    for (int cc = 0; cc < 4; ++cc) {
        const int jc = wv * 128 + cc * 32;   // K-chunk base (columns of P / rows of V)
        // Prefetch next chunk's V rows
        {
            const int jp = wv * 128 + ((cc + 1) & 3) * 32 + half * 16;
            __builtin_prefetch(vh + (size_t)jp * D_ + l16 * 4, 0, 3);
        }
        // A = P tile (16 rows x K=32) gathered from LDS, converted to bf16
        v16bf pa{};
        const float* prow = sc + l16 * S_ + jc;
        {
            const int kb = half * 16;
#pragma unroll
            for (int t = 0; t < 8; ++t) {
                pa[t]     = (__bf16)prow[kb + t];
                pa[t + 8] = (__bf16)prow[kb + 8 + t];
            }
        }
#pragma unroll
        for (int nt = 0; nt < 4; ++nt) {
            // B = V chunk (K=32 rows x N=16 cols): lane = N = d-column
            const int n = nt * 16 + l16;
            v16bf vb{};
#pragma unroll
            for (int t = 0; t < 16; ++t) {
                const int j = jc + half * 16 + t;
                vb[t] = (__bf16)vh[(size_t)j * D_ + n];
            }
            oacc[nt] = __builtin_amdgcn_wmma_f32_16x16x32_bf16(false, pa, false, vb,
                                                               (short)0, oacc[nt],
                                                               false, false);
        }
    }
    // partials to LDS: part[wv][i][d]
#pragma unroll
    for (int nt = 0; nt < 4; ++nt)
#pragma unroll
        for (int e = 0; e < 8; ++e)
            part[(wv * 16 + half * 8 + e) * 64 + nt * 16 + l16] = oacc[nt][e];
    __syncthreads();

    // ---- Phase 5: reduce 8 wave partials, write out ----
    {
        float* obase = out + (head * S_ + (size_t)mt * 16) * D_;
        for (int idx = tid; idx < 16 * 64; idx += 256) {
            float s = 0.f;
#pragma unroll
            for (int w2 = 0; w2 < 8; ++w2) s += part[w2 * 16 * 64 + idx];
            obase[idx] = s;    // idx = i*64 + d matches [i][d] row-major
        }
    }
}

extern "C" void kernel_launch(void* const* d_in, const int* in_sizes, int n_in,
                              void* d_out, int out_size, void* d_ws, size_t ws_size,
                              hipStream_t stream) {
    (void)in_sizes; (void)n_in; (void)out_size; (void)d_ws; (void)ws_size;
    const float* q     = (const float*)d_in[0];
    const float* k     = (const float*)d_in[1];
    const float* v     = (const float*)d_in[2];
    const float* tprop = (const float*)d_in[3];   // ATT_DEBIASING == "temporal"
    // d_in[4] = stat_prop_enc (unused for temporal debiasing)
    const int*   mask  = (const int*)d_in[5];

    float* out   = (float*)d_out;                          // [B,H,S,D]
    float* pattn = out + (size_t)B_ * H_ * S_ * D_;        // [B,H,S,S]

    // Allow >64KB dynamic LDS (CDNA5 supports up to 320KB per workgroup)
    hipFuncSetAttribute((const void*)debiased_attn_kernel,
                        hipFuncAttributeMaxDynamicSharedMemorySize, LDS_TOTAL);

    dim3 grid(S_ / 16, H_, B_);
    dim3 block(256);
    debiased_attn_kernel<<<grid, block, LDS_TOTAL, stream>>>(q, k, v, tprop, mask,
                                                             out, pattn);
}
